// LinearAttention_33921651704594
// MI455X (gfx1250) — matrix operand order repaired and checked by reading, not compiled, not run on hardware
//
#include <hip/hip_runtime.h>

typedef __attribute__((ext_vector_type(16))) _Float16 v16h;
typedef __attribute__((ext_vector_type(8)))  _Float16 v8h;
typedef __attribute__((ext_vector_type(4)))  _Float16 v4h;
typedef __attribute__((ext_vector_type(8)))  float    v8f;

#define B_    4
#define L_    4096
#define D_    1024
#define H_    16
#define MTOT  (B_*L_)   // 16384

__device__ __forceinline__ v8f wmma16(v16h a, v16h b, v8f c) {
    return __builtin_amdgcn_wmma_f32_16x16x32_f16(false, a, false, b, (short)0, c, false, false);
}

// ---- CDNA5 async global->LDS copies (ASYNCcnt path), with safe fallback ----
#if defined(__gfx1250__) && __has_builtin(__builtin_amdgcn_global_load_async_to_lds_b128)
#define HAVE_ASYNC 1
#else
#define HAVE_ASYNC 0
#endif

typedef __attribute__((ext_vector_type(4))) int v4i_;
typedef __attribute__((ext_vector_type(2))) int v2i_;
typedef __attribute__((address_space(1))) v4i_ g_v4i;   // global int4
typedef __attribute__((address_space(3))) v4i_ l_v4i;   // LDS int4
typedef __attribute__((address_space(1))) v2i_ g_v2i;   // global int2
typedef __attribute__((address_space(3))) v2i_ l_v2i;   // LDS int2

__device__ __forceinline__ void copy_b128(const _Float16* g, _Float16* l) {
#if HAVE_ASYNC
    __builtin_amdgcn_global_load_async_to_lds_b128((g_v4i*)g, (l_v4i*)l, 0, 0);
#else
    *(v8h*)l = *(const v8h*)g;
#endif
}
__device__ __forceinline__ void copy_b64(const _Float16* g, _Float16* l) {
#if HAVE_ASYNC
    __builtin_amdgcn_global_load_async_to_lds_b64((g_v2i*)g, (l_v2i*)l, 0, 0);
#else
    *(v4h*)l = *(const v4h*)g;
#endif
}
__device__ __forceinline__ void wait_async_le2() {
#if defined(__gfx1250__)
    asm volatile("s_wait_asynccnt 0x2" ::: "memory");
#endif
}
__device__ __forceinline__ void wait_async_0() {
#if defined(__gfx1250__)
    asm volatile("s_wait_asynccnt 0x0" ::: "memory");
#endif
}

// ---------------------------------------------------------------------------
// Prep: f32 -> f16 bulk convert (x, W_o), and f16 transpose for W_q/k/v.
// ---------------------------------------------------------------------------
__global__ __launch_bounds__(256)
void conv_f16(const float* __restrict__ in, _Float16* __restrict__ out)
{
    const long i = ((long)blockIdx.x * 256 + threadIdx.x) * 8;
    const float4 a = *(const float4*)(in + i);
    const float4 b = *(const float4*)(in + i + 4);
    v8h o;
    o[0] = (_Float16)a.x; o[1] = (_Float16)a.y; o[2] = (_Float16)a.z; o[3] = (_Float16)a.w;
    o[4] = (_Float16)b.x; o[5] = (_Float16)b.y; o[6] = (_Float16)b.z; o[7] = (_Float16)b.w;
    *(v8h*)(out + i) = o;
}

// Wt[n][k] = (f16) W[k][n]   (1024x1024), 32x32 tiles
__global__ __launch_bounds__(256)
void transpose_f16(const float* __restrict__ W, _Float16* __restrict__ Wt)
{
    __shared__ __attribute__((aligned(16))) _Float16 tile[32 * 36];
    const int t  = threadIdx.x;
    const int k0 = blockIdx.x * 32, n0 = blockIdx.y * 32;
    {
        const int kk = t >> 3, nc = (t & 7) * 4;
        const float4 v = *(const float4*)(W + (long)(k0 + kk) * D_ + n0 + nc);
        tile[(nc + 0) * 36 + kk] = (_Float16)v.x;
        tile[(nc + 1) * 36 + kk] = (_Float16)v.y;
        tile[(nc + 2) * 36 + kk] = (_Float16)v.z;
        tile[(nc + 3) * 36 + kk] = (_Float16)v.w;
    }
    __syncthreads();
    {
        const int nn = t >> 3, kc = (t & 7) * 4;
        v4h o;
        #pragma unroll
        for (int j = 0; j < 4; ++j) o[j] = tile[nn * 36 + kc + j];
        *(v4h*)(Wt + (long)(n0 + nn) * D_ + k0 + kc) = o;
    }
}

// ---------------------------------------------------------------------------
// f16 GEMM:  C[M,1024] = A[M,1024] * Bt^T   (Bt stored [n][k])
// block 256 = 8 waves, tile 64(M) x 32(N), K-step 32, double-buffered
// async-to-LDS pipeline.  MODE: 0=Q(phi) 1=K(phi*mask) 2=V(mask) 3=out(f32)
// ---------------------------------------------------------------------------
template<int MODE>
__global__ __launch_bounds__(256)
void gemm_f16(const _Float16* __restrict__ A, const _Float16* __restrict__ Bt,
              const unsigned char* __restrict__ mask,
              _Float16* __restrict__ Ch, float* __restrict__ Cf)
{
    __shared__ __attribute__((aligned(16))) _Float16 As[2][64 * 40];  // [m][k]
    __shared__ __attribute__((aligned(16))) _Float16 Bs[2][32 * 40];  // [n][k]

    const int t    = threadIdx.x;
    const int lane = t & 31;
    const int w    = t >> 5;
    const int wm   = w >> 1, wn = w & 1;
    const int m0   = blockIdx.x * 64;
    const int n0   = blockIdx.y * 32;
    const int l15  = lane & 15;
    const int hi   = (lane < 16) ? 0 : 1;

    // staging map: every thread issues exactly one b128 (A) + one b64 (B)
    // => every wave issues exactly 2 async instructions per tile.
    const int ar = t >> 2, ac = (t & 3) * 8;   // A: row 0..63, k-chunk of 8
    const int br = t >> 3, bc = (t & 7) * 4;   // B: n   0..31, k-chunk of 4
    const _Float16* Agp = A  + (long)(m0 + ar) * D_ + ac;
    const _Float16* Bgp = Bt + (long)(n0 + br) * D_ + bc;
    _Float16* Asd[2] = { &As[0][ar * 40 + ac], &As[1][ar * 40 + ac] };
    _Float16* Bsd[2] = { &Bs[0][br * 40 + bc], &Bs[1][br * 40 + bc] };

    // prologue: tile 0 in flight
    copy_b128(Agp, Asd[0]);
    copy_b64 (Bgp, Bsd[0]);

    v8f acc = {};

    for (int it = 0; it < 32; ++it) {
        const int cur = it & 1;
        if (it < 31) {                       // issue next tile, keep it in flight
            const int k0n = (it + 1) * 32;
            copy_b128(Agp + k0n, Asd[cur ^ 1]);
            copy_b64 (Bgp + k0n, Bsd[cur ^ 1]);
            wait_async_le2();                // current tile (2 older instrs) done
        } else {
            wait_async_0();
        }
        __syncthreads();                     // all waves' tile data visible

        // A fragment (ISA 16-bit A layout: two 8-half runs)
        const int mrow = wm * 16 + l15;
        const int kbA  = hi * 8;
        v8h alo = *(const v8h*)&As[cur][mrow * 40 + kbA];
        v8h ahi = *(const v8h*)&As[cur][mrow * 40 + kbA + 16];
        v16h a;
        #pragma unroll
        for (int j = 0; j < 8; ++j) { a[j] = alo[j]; a[8 + j] = ahi[j]; }

        // B fragment (ISA 16-bit layout: two 8-half runs, same as A)
        const int ncol = wn * 16 + l15;
        const int kbB  = hi * 8;
        v8h blo = *(const v8h*)&Bs[cur][ncol * 40 + kbB];
        v8h bhi = *(const v8h*)&Bs[cur][ncol * 40 + kbB + 16];
        v16h bf;
        #pragma unroll
        for (int j = 0; j < 8; ++j) { bf[j] = blo[j]; bf[8 + j] = bhi[j]; }

        acc = wmma16(a, bf, acc);
        __syncthreads();                     // reads done; next iter may overwrite
    }

    // epilogue in WMMA D layout
    const int gn = n0 + wn * 16 + l15;
    #pragma unroll
    for (int v = 0; v < 8; ++v) {
        const int gm = m0 + wm * 16 + v + hi * 8;
        float val = acc[v];
        if (MODE == 0 || MODE == 1)                 // phi = elu(x)+1
            val = (val > 0.f) ? (val + 1.f) : __expf(val);
        if (MODE == 1 || MODE == 2) {               // key padding mask
            const int b = gm >> 12, l = gm & (L_ - 1);
            val *= mask[b * L_ + l] ? 0.f : 1.f;
        }
        if (MODE == 3) Cf[(long)gm * D_ + gn] = val;
        else           Ch[(long)gm * D_ + gn] = (_Float16)val;
    }
}

// ---------------------------------------------------------------------------
// KV pooling: per (b,h): S_V[d][r] = sum_l V[l,d]*Kf[l,r], S_K[r]=sum_l Kf[l,r]
// 1 block per (b,h), 512 threads = 16 waves, one 16x16 S_V tile per wave.
// ---------------------------------------------------------------------------
__global__ __launch_bounds__(512)
void pool_kernel(const _Float16* __restrict__ Kf, const _Float16* __restrict__ Vm,
                 float* __restrict__ sv, float* __restrict__ sk)
{
    __shared__ __attribute__((aligned(16))) _Float16 VsT[64 * 40];  // [d][l]
    __shared__ __attribute__((aligned(16))) _Float16 KsT[64 * 40];  // [r][l]

    const int bh = blockIdx.x;
    const int b = bh >> 4, h = bh & 15;
    const int t    = threadIdx.x;
    const int lane = t & 31;
    const int w    = t >> 5;
    const int dt   = w >> 2, rt = w & 3;
    const int l15  = lane & 15;
    const int hi   = (lane < 16) ? 0 : 1;
    const int ll   = t >> 4;              // l within chunk
    const int cc   = (t & 15) * 4;        // col within 64

    v8f acc = {};
    float skacc = 0.f;

    for (int l0 = 0; l0 < L_; l0 += 32) {
        const long rowoff = ((long)(b * L_ + l0 + ll)) * D_ + h * 64 + cc;
        v4h kv = *(const v4h*)(Kf + rowoff);
        v4h vv = *(const v4h*)(Vm + rowoff);
        if (l0 + 32 < L_) {                           // stream-ahead hints
            __builtin_prefetch(Kf + rowoff + 32 * D_, 0, 3);
            __builtin_prefetch(Vm + rowoff + 32 * D_, 0, 3);
        }
        #pragma unroll
        for (int j = 0; j < 4; ++j) {
            KsT[(cc + j) * 40 + ll] = kv[j];
            VsT[(cc + j) * 40 + ll] = vv[j];
        }
        __syncthreads();

        const int drow = dt * 16 + l15;
        const int kbA  = hi * 8;
        v8h alo = *(const v8h*)&VsT[drow * 40 + kbA];
        v8h ahi = *(const v8h*)&VsT[drow * 40 + kbA + 16];
        v16h a;
        #pragma unroll
        for (int j = 0; j < 8; ++j) { a[j] = alo[j]; a[8 + j] = ahi[j]; }

        const int rcol = rt * 16 + l15;
        const int kbB  = hi * 8;
        v8h blo = *(const v8h*)&KsT[rcol * 40 + kbB];
        v8h bhi = *(const v8h*)&KsT[rcol * 40 + kbB + 16];
        v16h bf;
        #pragma unroll
        for (int j = 0; j < 8; ++j) { bf[j] = blo[j]; bf[8 + j] = bhi[j]; }

        acc = wmma16(a, bf, acc);

        if (t < 64) {                      // S_K side reduction (r = t)
            #pragma unroll
            for (int j = 0; j < 32; ++j) skacc += (float)KsT[t * 40 + j];
        }
        __syncthreads();
    }

    #pragma unroll
    for (int v = 0; v < 8; ++v) {
        const int d = dt * 16 + v + hi * 8;
        const int r = rt * 16 + l15;
        sv[((long)bh * 64 + d) * 64 + r] = acc[v];
    }
    if (t < 64) sk[(long)bh * 64 + t] = skacc;
}

// ---------------------------------------------------------------------------
// Recombine: out[l,d] = (sum_r Qf[l,r]*S_V[d,r]) / (sum_r Qf[l,r]*S_K[r]+eps)
// ---------------------------------------------------------------------------
__global__ __launch_bounds__(256)
void attn_kernel(const _Float16* __restrict__ Qf, const float* __restrict__ sv,
                 const float* __restrict__ sk, _Float16* __restrict__ attn)
{
    const int t = threadIdx.x, lane = t & 31, w = t >> 5;
    const int h = blockIdx.y, b = blockIdx.z;
    const int l0  = (blockIdx.x * 8 + w) * 16;
    const int l15 = lane & 15;
    const int hi  = (lane < 16) ? 0 : 1;

    const long qrow = ((long)(b * L_ + l0 + l15)) * D_ + h * 64;

    v16h a[2];
    #pragma unroll
    for (int s = 0; s < 2; ++s) {
        v8h lo = *(const v8h*)(Qf + qrow + s * 32 + hi * 8);
        v8h hh = *(const v8h*)(Qf + qrow + s * 32 + hi * 8 + 16);
        #pragma unroll
        for (int j = 0; j < 8; ++j) { a[s][j] = lo[j]; a[s][8 + j] = hh[j]; }
    }

    const float* skp = sk + (long)(b * H_ + h) * 64;
    float den = 0.f;
    const int rb = hi * 32;
    #pragma unroll
    for (int r = 0; r < 32; ++r)
        den += (float)Qf[qrow + rb + r] * skp[rb + r];
    den += __shfl_xor(den, 16, 32);

    const float* svp0 = sv + (long)(b * H_ + h) * 64 * 64;

    #pragma unroll
    for (int dtile = 0; dtile < 4; ++dtile) {
        v8f acc = {};
        const int d = dtile * 16 + l15;
        #pragma unroll
        for (int s = 0; s < 2; ++s) {
            const float* svp = svp0 + (long)d * 64 + s * 32 + hi * 8;
            v16h bf;
            #pragma unroll
            for (int j = 0; j < 16; ++j) bf[j] = (_Float16)svp[j < 8 ? j : j + 8];
            acc = wmma16(a[s], bf, acc);
        }
        #pragma unroll
        for (int v = 0; v < 8; ++v) {
            const int M = v + hi * 8;
            const float dv = __shfl(den, M, 32);
            const float o  = acc[v] / (dv + 1e-6f);
            attn[((long)(b * L_ + l0 + M)) * D_ + h * 64 + dtile * 16 + l15] = (_Float16)o;
        }
    }
}

// ---------------------------------------------------------------------------
extern "C" void kernel_launch(void* const* d_in, const int* in_sizes, int n_in,
                              void* d_out, int out_size, void* d_ws, size_t ws_size,
                              hipStream_t stream)
{
    (void)in_sizes; (void)n_in; (void)out_size; (void)ws_size;
    const float* x            = (const float*)d_in[0];
    const unsigned char* mask = (const unsigned char*)d_in[1];   // True = PAD
    const float* Wq           = (const float*)d_in[2];
    const float* Wk           = (const float*)d_in[3];
    const float* Wv           = (const float*)d_in[4];
    const float* Wo           = (const float*)d_in[5];

    // workspace
    _Float16* xh  = (_Float16*)d_ws;                 // 16M
    _Float16* Qf  = xh  + (long)MTOT * D_;
    _Float16* Kf  = Qf  + (long)MTOT * D_;
    _Float16* Vm  = Kf  + (long)MTOT * D_;
    _Float16* At  = Vm  + (long)MTOT * D_;
    _Float16* Wqt = At  + (long)MTOT * D_;           // 1M each
    _Float16* Wkt = Wqt + (long)D_ * D_;
    _Float16* Wvt = Wkt + (long)D_ * D_;
    _Float16* Wot = Wvt + (long)D_ * D_;
    float*    sv  = (float*)(Wot + (long)D_ * D_);   // 64*64*64
    float*    sk  = sv + 64 * 64 * 64;               // 64*64

    // prep: conversions + weight transposes (W_o is used as B(k,n)=Wo[n][k],
    // which is already the [n][k] layout -> plain convert)
    conv_f16<<<dim3((MTOT * D_) / 2048), dim3(256), 0, stream>>>(x, xh);
    dim3 tg(32, 32);
    transpose_f16<<<tg, dim3(256), 0, stream>>>(Wq, Wqt);
    transpose_f16<<<tg, dim3(256), 0, stream>>>(Wk, Wkt);
    transpose_f16<<<tg, dim3(256), 0, stream>>>(Wv, Wvt);
    conv_f16<<<dim3((D_ * D_) / 2048), dim3(256), 0, stream>>>(Wo, Wot);

    dim3 gg(MTOT / 64, D_ / 32), bb(256);
    gemm_f16<0><<<gg, bb, 0, stream>>>(xh, Wqt, mask, Qf, nullptr);
    gemm_f16<1><<<gg, bb, 0, stream>>>(xh, Wkt, mask, Kf, nullptr);
    gemm_f16<2><<<gg, bb, 0, stream>>>(xh, Wvt, mask, Vm, nullptr);

    pool_kernel<<<dim3(B_ * H_), dim3(512), 0, stream>>>(Kf, Vm, sv, sk);

    attn_kernel<<<dim3(L_ / 128, H_, B_), dim3(256), 0, stream>>>(Qf, sv, sk, At);

    gemm_f16<3><<<gg, bb, 0, stream>>>(At, Wot, nullptr, nullptr, (float*)d_out);
}
